// BertFinetun_26800595927385
// MI455X (gfx1250) — matrix operand order repaired and checked by reading, not compiled, not run on hardware
//
#include <hip/hip_runtime.h>
#include <hip/hip_bf16.h>
#include <stdint.h>

typedef __attribute__((ext_vector_type(16))) _Float16 v16h;
typedef __attribute__((ext_vector_type(8)))  _Float16 v8h;
typedef __attribute__((ext_vector_type(8)))  float    v8f;

#define BB   64
#define SS   512
#define HH   768
#define FDD  74
#define PPAD 32
#define ROWH 40   // LDS row stride in halfs (64B data + 16B TDM pad = 80B, conflict-free)

static constexpr size_t OFF_H = 0;
static constexpr size_t OFF_T = (size_t)BB * HH;                 // 49152
static constexpr size_t OFF_F = OFF_T + (size_t)BB * SS * SS;    // 16826368

// ---------------------------------------------------------------------------
// LDS gather of one WMMA 16-half fragment piece (two 16B chunks).
// Rows are 80B apart (TDM-padded), so lanes 0..15 land on distinct banks.
// ---------------------------------------------------------------------------
__device__ inline v16h lds_gather(const _Float16* base, int row, int ca, int cb) {
    const v8h* p = (const v8h*)(base + row * ROWH);
    v8h x = p[ca];
    v8h y = p[cb];
    v16h r;
#pragma unroll
    for (int e = 0; e < 8; ++e) { r[e] = x[e]; r[8 + e] = y[e]; }
    return r;
}

// ---------------------------------------------------------------------------
// TDM: async-load a contiguous (16 dwords x 512 rows) tile into LDS,
// inserting 4 dwords of padding after every 16 dwords (80B LDS row stride).
// D# fields per CDNA5 ISA 8.3/8.4 (group0: count/lds/global/type,
// group1: data_size, pad, dims, strides). Groups 2/3 zero (2D tensor).
// ---------------------------------------------------------------------------
#if __has_builtin(__builtin_amdgcn_tensor_load_to_lds) && __has_builtin(__builtin_amdgcn_s_wait_tensorcnt)
#define USE_TDM 1
#endif

#if defined(USE_TDM)
__device__ inline void tdm_load_2d_pad(uint32_t lds_off, const void* gsrc) {
    typedef __attribute__((ext_vector_type(4))) unsigned int u32x4;
    typedef __attribute__((ext_vector_type(4))) int i32x4;
    typedef __attribute__((ext_vector_type(8))) int i32x8;
    const uint64_t ga = (uint64_t)(uintptr_t)gsrc;

    u32x4 g0;
    g0[0] = 1u;                                                 // count=1, user mode
    g0[1] = lds_off;                                            // lds_addr (bytes)
    g0[2] = (uint32_t)ga;                                       // global_addr[95:64]
    g0[3] = (uint32_t)((ga >> 32) & 0x01ffffffu) | (2u << 30);  // addr[120:96] | type=2

    const uint32_t td0 = 16, td1 = SS, tile0 = 16, tile1 = SS;  // dword units
    const uint64_t s0 = 16;                                     // dim0 stride (contiguous)
    // data_size=4B(code2), pad_enable, pad_interval=16 dwords(code3), pad_amount=4 dwords(code3)
    const uint32_t w0 = (2u << 16) | (1u << 20) | (3u << 22) | (3u << 25);
    const uint64_t q0 = (uint64_t)w0 | ((uint64_t)(td0 & 0xffffu) << 48);
    const uint64_t q1 = (uint64_t)(td0 >> 16) | ((uint64_t)td1 << 16) | ((uint64_t)tile0 << 48);
    const uint64_t q2 = (uint64_t)tile1 | (s0 << 32);           // tile_dim2 = 0
    const uint64_t q3 = (s0 >> 32) & 0xffffu;                   // dim1 stride = 0 (unused, 2D)

    i32x8 g1;
    g1[0] = (int)(uint32_t)q0;  g1[1] = (int)(uint32_t)(q0 >> 32);
    g1[2] = (int)(uint32_t)q1;  g1[3] = (int)(uint32_t)(q1 >> 32);
    g1[4] = (int)(uint32_t)q2;  g1[5] = (int)(uint32_t)(q2 >> 32);
    g1[6] = (int)(uint32_t)q3;  g1[7] = (int)(uint32_t)(q3 >> 32);

    i32x4 z4 = {0, 0, 0, 0};
#if defined(__clang_major__) && (__clang_major__ >= 23)
    i32x8 z8 = {0, 0, 0, 0, 0, 0, 0, 0};
    __builtin_amdgcn_tensor_load_to_lds(g0, g1, z4, z4, z8, 0);
#else
    __builtin_amdgcn_tensor_load_to_lds(g0, g1, z4, z4, 0);
#endif
}
#endif

// ---------------------------------------------------------------------------
// Projection: dst(BS x 32 f16) = src(BS x KSRC f32) @ W(30 x KSRC)^T, pad cols = 0
// One wave per 16-row M tile, N=32 (two accumulators), K loop of KT x 32.
// ---------------------------------------------------------------------------
template <int KSRC, int KT>
__global__ __launch_bounds__(256) void proj_kernel(const float* __restrict__ src,
                                                   const float* __restrict__ W,
                                                   _Float16* __restrict__ dst,
                                                   float* __restrict__ sumsq,
                                                   int do_sumsq) {
    const int lane = threadIdx.x & 31;
    const int wave = threadIdx.x >> 5;
    const int rowbase = (blockIdx.x * 8 + wave) * 16;
    const int rl = lane & 15;
    const int kh = (lane < 16) ? 0 : 8;
    const int kb = (lane < 16) ? 0 : 16;
    const int n0 = rl;
    const int n1 = rl + 16;

    const float* arow_p = src + (size_t)(rowbase + rl) * KSRC;
    const float* b0_p = (n0 < 30) ? (W + (size_t)n0 * KSRC) : nullptr;
    const float* b1_p = (n1 < 30) ? (W + (size_t)n1 * KSRC) : nullptr;

    v8f c0 = {}; v8f c1 = {};
    for (int kt = 0; kt < KT; ++kt) {
        const int k0 = kt * 32;
        v16h a, b0, b1;
#pragma unroll
        for (int e = 0; e < 8; ++e) {
            int ka0 = k0 + kb + e;
            int ka1 = k0 + kb + 8 + e;
            a[e]     = (_Float16)((ka0 < KSRC) ? arow_p[ka0] : 0.f);
            a[8 + e] = (_Float16)((ka1 < KSRC) ? arow_p[ka1] : 0.f);
        }
#pragma unroll
        for (int e = 0; e < 16; ++e) {
            int kk = k0 + kb + e;
            b0[e] = (_Float16)((b0_p && kk < KSRC) ? b0_p[kk] : 0.f);
            b1[e] = (_Float16)((b1_p && kk < KSRC) ? b1_p[kk] : 0.f);
        }
        c0 = __builtin_amdgcn_wmma_f32_16x16x32_f16(false, a, false, b0, (short)0, c0, false, false);
        c1 = __builtin_amdgcn_wmma_f32_16x16x32_f16(false, a, false, b1, (short)0, c1, false, false);
    }

    float lsum = 0.f;
    const int mrow0 = rowbase + ((lane < 16) ? 0 : 8);
#pragma unroll
    for (int r = 0; r < 8; ++r) {
        int row = mrow0 + r;
        float v0 = c0[r], v1 = c1[r];
        lsum += v0 * v0 + v1 * v1;   // pad columns are exactly 0
        dst[(size_t)row * PPAD + n0] = (_Float16)v0;
        dst[(size_t)row * PPAD + n1] = (_Float16)v1;
    }
    if (do_sumsq) {
#pragma unroll
        for (int off = 16; off > 0; off >>= 1) lsum += __shfl_down(lsum, off, 32);
        if (lane == 0) atomicAdd(sumsq, lsum);
    }
}

__global__ void scale_kernel(const float* __restrict__ sumsq, float* __restrict__ scale) {
    *scale = rsqrtf(*sumsq);   // 1/w^2 = (sum t^2)^(-1/2)
}

// ---------------------------------------------------------------------------
// Attention kernel: writes text_att1 and fusion_att1 (store-BW bound).
// Block = 8 waves; each wave owns one 16-row tile x all 512 cols.
// Batch's text+audio rows staged once into 80KB dynamic LDS via TDM.
// ---------------------------------------------------------------------------
__global__ __launch_bounds__(256) void attn_kernel(const _Float16* __restrict__ gt,
                                                   const _Float16* __restrict__ ga,
                                                   const float* __restrict__ scale_p,
                                                   const float* __restrict__ tw_p,
                                                   const float* __restrict__ aw_p,
                                                   const float* __restrict__ fb_p,
                                                   float* __restrict__ outT,
                                                   float* __restrict__ outF) {
    extern __shared__ _Float16 smem[];           // 2 * SS * ROWH halfs = 80KB
    _Float16* lt = smem;                          // dynamic LDS base offset 0
    _Float16* la = smem + SS * ROWH;              // byte offset 40960

    const int b    = blockIdx.x >> 2;
    const int seg  = blockIdx.x & 3;
    const int lane = threadIdx.x & 31;
    const int wave = threadIdx.x >> 5;

#if defined(USE_TDM)
    if (threadIdx.x < 32) {                       // wave 0 drives the TDM
        tdm_load_2d_pad(0u, gt + (size_t)b * SS * PPAD);
        tdm_load_2d_pad((uint32_t)(SS * ROWH * sizeof(_Float16)), ga + (size_t)b * SS * PPAD);
        __builtin_amdgcn_s_wait_tensorcnt(0);
    }
#elif __has_builtin(__builtin_amdgcn_global_load_async_to_lds_b128)
    {
        const uint4* srcT = (const uint4*)(gt + (size_t)b * SS * PPAD);
        const uint4* srcA = (const uint4*)(ga + (size_t)b * SS * PPAD);
        uint4* dT = (uint4*)lt;
        uint4* dA = (uint4*)la;
        for (int idx = threadIdx.x; idx < SS * 4; idx += 256) {
            int r = idx >> 2, c = idx & 3;
            __builtin_amdgcn_global_load_async_to_lds_b128(
                (__attribute__((address_space(1))) const void*)(srcT + idx),
                (__attribute__((address_space(3))) void*)(dT + r * 5 + c), 0, 0);
            __builtin_amdgcn_global_load_async_to_lds_b128(
                (__attribute__((address_space(1))) const void*)(srcA + idx),
                (__attribute__((address_space(3))) void*)(dA + r * 5 + c), 0, 0);
        }
#if __has_builtin(__builtin_amdgcn_s_wait_asynccnt)
        __builtin_amdgcn_s_wait_asynccnt(0);
#else
        asm volatile("s_wait_asynccnt 0" ::: "memory");
#endif
    }
#else
    {
        const uint4* srcT = (const uint4*)(gt + (size_t)b * SS * PPAD);
        const uint4* srcA = (const uint4*)(ga + (size_t)b * SS * PPAD);
        uint4* dT = (uint4*)lt;
        uint4* dA = (uint4*)la;
        for (int idx = threadIdx.x; idx < SS * 4; idx += 256) {
            int r = idx >> 2, c = idx & 3;
            dT[r * 5 + c] = srcT[idx];
            dA[r * 5 + c] = srcA[idx];
        }
    }
#endif
    __syncthreads();

    const float s  = *scale_p;
    const float tw = *tw_p, aw = *aw_p, fb = *fb_p;

    const int rl = lane & 15;
    const int hi = lane >> 4;
    const int arow = seg * 128 + wave * 16 + rl;

    const v16h At = lds_gather(lt, arow, 2 * hi, 2 * hi + 1);
    const v16h Aa = lds_gather(la, arow, 2 * hi, 2 * hi + 1);

    for (int j = 0; j < 32; ++j) {
        const int brow = j * 16 + rl;
        v16h Bt = lds_gather(lt, brow, 2 * hi, 2 * hi + 1);
        v16h Ba = lds_gather(la, brow, 2 * hi, 2 * hi + 1);
        v8f zz = {};
        v8f tc = __builtin_amdgcn_wmma_f32_16x16x32_f16(false, At, false, Bt, (short)0, zz, false, false);
        v8f ac = __builtin_amdgcn_wmma_f32_16x16x32_f16(false, Aa, false, Ba, (short)0, zz, false, false);
        const int col  = j * 16 + rl;
        const int rowb = seg * 128 + wave * 16 + (hi ? 8 : 0);
#pragma unroll
        for (int r = 0; r < 8; ++r) {
            int row = rowb + r;
            size_t o = ((size_t)b * SS + row) * SS + col;
            float tv = fmaxf(tc[r] * s, 0.f);              // text_att1
            float av = fmaxf(ac[r], 0.f);                  // audio_att
            float fv = fmaxf(tw * tv + aw * av + fb, 0.f); // fusion_att1
            outT[o] = tv;
            outF[o] = fv;
        }
    }
}

// ---------------------------------------------------------------------------
// Head kernel: only row 0 per batch needs softmax -> fusion_data -> dense -> LN.
// ---------------------------------------------------------------------------
__global__ __launch_bounds__(256) void head_kernel(const float* __restrict__ hidden,
                                                   const float* __restrict__ mask,
                                                   const _Float16* __restrict__ gt,
                                                   const _Float16* __restrict__ ga,
                                                   const float* __restrict__ scale_p,
                                                   const float* __restrict__ tw_p,
                                                   const float* __restrict__ aw_p,
                                                   const float* __restrict__ fb_p,
                                                   const float* __restrict__ W,
                                                   const float* __restrict__ bias,
                                                   const float* __restrict__ lnw,
                                                   const float* __restrict__ lnb,
                                                   float* __restrict__ outH) {
    __shared__ float fv[SS];
    __shared__ float fd[HH];
    __shared__ float ho[HH];
    __shared__ float red[256];
    __shared__ float t0s[PPAD], a0s[PPAD];

    const int b = blockIdx.x;
    const int tid = threadIdx.x;
    const float s = *scale_p, tw = *tw_p, aw = *aw_p, fb = *fb_p;

    if (tid < PPAD) {
        t0s[tid] = (float)gt[(size_t)b * SS * PPAD + tid];
        a0s[tid] = (float)ga[(size_t)b * SS * PPAD + tid];
    }
    __syncthreads();

    for (int j = tid; j < SS; j += 256) {
        const _Float16* tj = gt + ((size_t)b * SS + j) * PPAD;
        const _Float16* aj = ga + ((size_t)b * SS + j) * PPAD;
        float td = 0.f, ad = 0.f;
#pragma unroll
        for (int k = 0; k < PPAD; ++k) {
            td += t0s[k] * (float)tj[k];
            ad += a0s[k] * (float)aj[k];
        }
        float f = tw * fmaxf(td * s, 0.f) + aw * fmaxf(ad, 0.f) + fb;
        fv[j] = f + mask[(size_t)b * SS + j] + mask[(size_t)b * SS];
    }
    __syncthreads();

    float m = -INFINITY;
    for (int j = tid; j < SS; j += 256) m = fmaxf(m, fv[j]);
    red[tid] = m; __syncthreads();
    for (int st = 128; st > 0; st >>= 1) { if (tid < st) red[tid] = fmaxf(red[tid], red[tid + st]); __syncthreads(); }
    m = red[0]; __syncthreads();
    float ls = 0.f;
    for (int j = tid; j < SS; j += 256) { float e = __expf(fv[j] - m); fv[j] = e; ls += e; }
    red[tid] = ls; __syncthreads();
    for (int st = 128; st > 0; st >>= 1) { if (tid < st) red[tid] += red[tid + st]; __syncthreads(); }
    const float inv = 1.f / red[0]; __syncthreads();
    for (int j = tid; j < SS; j += 256) fv[j] *= inv;
    __syncthreads();

    for (int h = tid; h < HH; h += 256) {
        const float* hp = hidden + (size_t)b * SS * HH + h;
        float acc = hp[0];
        for (int j = 0; j < SS; ++j) acc += fv[j] * hp[(size_t)j * HH];
        fd[h] = acc;
    }
    __syncthreads();

    for (int o = tid; o < HH; o += 256) {
        const float* wr = W + (size_t)o * HH;
        float acc = bias[o];
        for (int h = 0; h < HH; ++h) acc += fd[h] * wr[h];
        ho[o] = acc;
    }
    __syncthreads();

    float ps = 0.f;
    for (int o = tid; o < HH; o += 256) ps += ho[o];
    red[tid] = ps; __syncthreads();
    for (int st = 128; st > 0; st >>= 1) { if (tid < st) red[tid] += red[tid + st]; __syncthreads(); }
    const float u = red[0] / (float)HH; __syncthreads();
    float pv = 0.f;
    for (int o = tid; o < HH; o += 256) { float d = ho[o] - u; pv += d * d; }
    red[tid] = pv; __syncthreads();
    for (int st = 128; st > 0; st >>= 1) { if (tid < st) red[tid] += red[tid + st]; __syncthreads(); }
    const float rinv = rsqrtf(red[0] / (float)HH + 1e-12f);
    for (int o = tid; o < HH; o += 256)
        outH[(size_t)b * HH + o] = lnw[o] * (ho[o] - u) * rinv + lnb[o];
}

// ---------------------------------------------------------------------------
extern "C" void kernel_launch(void* const* d_in, const int* in_sizes, int n_in,
                              void* d_out, int out_size, void* d_ws, size_t ws_size,
                              hipStream_t stream) {
    (void)in_sizes; (void)n_in; (void)out_size; (void)ws_size;
    const float* hidden = (const float*)d_in[0];
    const float* audio  = (const float*)d_in[1];
    const float* mask   = (const float*)d_in[2];
    const float* Wt     = (const float*)d_in[3];
    const float* Wa     = (const float*)d_in[4];
    const float* tw     = (const float*)d_in[5];
    const float* aw     = (const float*)d_in[6];
    const float* fb     = (const float*)d_in[7];
    const float* dW     = (const float*)d_in[8];
    const float* db     = (const float*)d_in[9];
    const float* lnw    = (const float*)d_in[10];
    const float* lnb    = (const float*)d_in[11];
    float* out = (float*)d_out;

    float* sumsq = (float*)d_ws;
    float* scale = sumsq + 1;
    _Float16* t16 = (_Float16*)((char*)d_ws + 256);
    _Float16* a16 = t16 + (size_t)BB * SS * PPAD;

    hipMemsetAsync(d_ws, 0, 8, stream);

    dim3 blk(256);
    proj_kernel<HH, 24><<<dim3(BB * SS / 128), blk, 0, stream>>>(hidden, Wt, t16, sumsq, 1);
    proj_kernel<FDD, 3><<<dim3(BB * SS / 128), blk, 0, stream>>>(audio, Wa, a16, sumsq, 0);
    scale_kernel<<<1, 1, 0, stream>>>(sumsq, scale);
    attn_kernel<<<dim3(BB * 4), blk, (size_t)(2 * SS * ROWH * sizeof(_Float16)), stream>>>(
        t16, a16, scale, tw, aw, fb, out + OFF_T, out + OFF_F);
    head_kernel<<<dim3(BB), blk, 0, stream>>>(hidden, mask, t16, a16, scale, tw, aw, fb,
                                              dW, db, lnw, lnb, out + OFF_H);
}
